// RetrievalAugmentedKinship_22204980920632
// MI455X (gfx1250) — compile-verified, matrix operand order checked by reading, not run on hardware
//
#include <hip/hip_runtime.h>

typedef __bf16 bf16;
typedef __attribute__((ext_vector_type(16))) __bf16 v16bf;
typedef __attribute__((ext_vector_type(8)))  __bf16 v8bf;
typedef __attribute__((ext_vector_type(8)))  float  v8f;
typedef __attribute__((ext_vector_type(4)))  float  v4f;

#define BQ      256
#define NG      196608
#define D3      1536
#define TOPK    32
#define CHUNK   4096
#define NCHUNK  48          // NG / CHUNK
#define KSTEPS  48          // D3 / 32
#define NTILES  256         // CHUNK / 16
#define QS      1544        // padded bf16 row stride for query tile (772 words -> 16 distinct banks)
#define SS      4100        // padded fp32 row stride for score board (conflict-free m-strided stores)

#define NEG_INF (-__builtin_inff())

extern __shared__ unsigned char smem_raw[];

__device__ __forceinline__ v16bf cvt_b_frag(const float* gp) {
    // 16 contiguous-K fp32 values in the WMMA 16-bit operand pattern:
    // elements 0..7 = K {k0..k0+7}, elements 8..15 = K {k0+16..k0+23}
    v4f b0 = *(const v4f*)(gp);
    v4f b1 = *(const v4f*)(gp + 4);
    v4f b2 = *(const v4f*)(gp + 16);
    v4f b3 = *(const v4f*)(gp + 20);
    v16bf b;
#pragma unroll
    for (int i = 0; i < 4; ++i) {
        b[i]      = (bf16)b0[i];
        b[i + 4]  = (bf16)b1[i];
        b[i + 8]  = (bf16)b2[i];
        b[i + 12] = (bf16)b3[i];
    }
    return b;
}

// ---------------------------------------------------------------------------
// Phase 1: block = (m-tile of 16 queries) x (one 4096-row gallery chunk).
// bf16 WMMA GEMM into an LDS score board, mask, then per-query online top-32.
// Each wave works on TWO n-tiles per k-step: 2 independent WMMA chains and
// one shared A fragment (halves the ds_load_b128 traffic per WMMA).
// ---------------------------------------------------------------------------
__global__ void __launch_bounds__(256)
sim_topk_phase1(const float* __restrict__ q,
                const float* __restrict__ g,
                const unsigned char* __restrict__ mask,
                float* __restrict__ ws_s,
                int*   __restrict__ ws_i)
{
    const int mt    = blockIdx.x;        // 0..15  (query tile)
    const int chunk = blockIdx.y;        // 0..47  (gallery chunk)
    const int tid   = threadIdx.x;
    const int lane  = tid & 31;
    const int wave  = tid >> 5;
    const int half  = lane >> 4;         // 0: K {0..7,16..23}, 1: K {8..15,24..31}
    const int l16   = lane & 15;

    bf16*  qs = (bf16*)smem_raw;                                 // [16][QS] bf16
    float* sc = (float*)(smem_raw + 16 * QS * sizeof(bf16));     // [16][SS] fp32

    // ---- stage 16 query rows fp32 -> bf16 into LDS (coalesced) ----
    for (int m = 0; m < 16; ++m) {
        const float* qrow = q + (size_t)(mt * 16 + m) * D3;
        for (int k = tid; k < D3; k += 256)
            qs[m * QS + k] = (bf16)qrow[k];
    }
    __syncthreads();

    // ---- GEMM: each wave owns n-tile pairs (2*p, 2*p+1), p = wave, wave+8,...
    for (int ntp = wave; ntp < NTILES / 2; ntp += 8) {
        const int nbase0 = (2 * ntp) * 16;
        const int nbase1 = nbase0 + 16;
        const size_t grow0 = (size_t)(chunk * CHUNK + nbase0 + l16) * D3;
        const size_t grow1 = (size_t)(chunk * CHUNK + nbase1 + l16) * D3;
        v8f c0 = {};
        v8f c1 = {};
        for (int kc = 0; kc < KSTEPS; ++kc) {
            const int k0 = kc * 32 + half * 8;

            // Shared A fragment: query row l16, K {k0..k0+7} and {k0+16..k0+23}
            v8bf a0 = *(const v8bf*)(qs + l16 * QS + k0);
            v8bf a1 = *(const v8bf*)(qs + l16 * QS + k0 + 16);
            v16bf a;
#pragma unroll
            for (int i = 0; i < 8; ++i) { a[i] = a0[i]; a[i + 8] = a1[i]; }

            // Two B fragments (gallery rows as columns), fp32 -> bf16
            v16bf b0 = cvt_b_frag(g + grow0 + k0);
            v16bf b1 = cvt_b_frag(g + grow1 + k0);

            c0 = __builtin_amdgcn_wmma_f32_16x16x32_bf16(
                     false, a, false, b0, (short)0, c0, false, false);
            c1 = __builtin_amdgcn_wmma_f32_16x16x32_bf16(
                     false, a, false, b1, (short)0, c1, false, false);
        }

        // C layout: VGPR i -> M = i + 8*half, N = l16. Apply mask, store to LDS.
        const size_t mcol0 = (size_t)chunk * CHUNK + nbase0 + l16;
        const size_t mcol1 = (size_t)chunk * CHUNK + nbase1 + l16;
#pragma unroll
        for (int i = 0; i < 8; ++i) {
            const int m = i + half * 8;
            const size_t mrow = (size_t)(mt * 16 + m) * NG;
            const unsigned char ex0 = mask[mrow + mcol0];
            const unsigned char ex1 = mask[mrow + mcol1];
            sc[m * SS + nbase0 + l16] = ex0 ? NEG_INF : c0[i];
            sc[m * SS + nbase1 + l16] = ex1 ? NEG_INF : c1[i];
        }
    }
    __syncthreads();

    // ---- per-query top-32 over the 4096-entry score row (wave argmax) ----
    for (int qq = wave; qq < 16; qq += 8) {
        float* row = sc + qq * SS;
        float lval = NEG_INF; int lidx = -1;
        for (int t = 0; t < CHUNK / 32; ++t) {
            const float v = row[lane + t * 32];
            if (v > lval) { lval = v; lidx = lane + t * 32; }
        }
        const int qg = mt * 16 + qq;
        for (int r = 0; r < TOPK; ++r) {
            float bv = lval; int bi = lidx;
#pragma unroll
            for (int off = 16; off; off >>= 1) {
                const float ov = __shfl_xor(bv, off, 32);
                const int   oi = __shfl_xor(bi, off, 32);
                if (ov > bv || (ov == bv && oi > bi)) { bv = ov; bi = oi; }
            }
            if (lane == 0) {
                ws_s[(size_t)qg * (NCHUNK * TOPK) + chunk * TOPK + r] = bv;
                ws_i[(size_t)qg * (NCHUNK * TOPK) + chunk * TOPK + r] =
                    (bi >= 0) ? (chunk * CHUNK + bi) : -1;
            }
            if (bi >= 0 && lidx == bi) {            // unique winner lane
                row[bi] = NEG_INF;
                lval = NEG_INF; lidx = -1;
                for (int t = 0; t < CHUNK / 32; ++t) {
                    const float v = row[lane + t * 32];
                    if (v > lval) { lval = v; lidx = lane + t * 32; }
                }
            }
        }
    }
}

// ---------------------------------------------------------------------------
// Phase 2: one wave per query merges 48*32 candidates into sorted top-32.
// ---------------------------------------------------------------------------
__global__ void __launch_bounds__(32)
topk_merge(const float* __restrict__ ws_s,
           const int*   __restrict__ ws_i,
           int*   __restrict__ out_i,
           float* __restrict__ out_s)
{
    __shared__ float cs[NCHUNK * TOPK];
    __shared__ int   ci[NCHUNK * TOPK];
    const int qg = blockIdx.x;
    const int lane = threadIdx.x;

    for (int t = lane; t < NCHUNK * TOPK; t += 32) {
        cs[t] = ws_s[(size_t)qg * (NCHUNK * TOPK) + t];
        ci[t] = ws_i[(size_t)qg * (NCHUNK * TOPK) + t];
    }
    __syncthreads();

    float lval = NEG_INF; int lidx = -1;
    for (int t = 0; t < NCHUNK; ++t) {
        const float v = cs[lane + t * 32];
        if (v > lval) { lval = v; lidx = lane + t * 32; }
    }
    for (int r = 0; r < TOPK; ++r) {
        float bv = lval; int bi = lidx;
#pragma unroll
        for (int off = 16; off; off >>= 1) {
            const float ov = __shfl_xor(bv, off, 32);
            const int   oi = __shfl_xor(bi, off, 32);
            if (ov > bv || (ov == bv && oi > bi)) { bv = ov; bi = oi; }
        }
        if (lane == 0) {
            out_i[qg * TOPK + r] = (bi >= 0) ? ci[bi] : -1;
            out_s[qg * TOPK + r] = bv;
        }
        if (bi >= 0 && lidx == bi) {
            cs[bi] = NEG_INF;
            lval = NEG_INF; lidx = -1;
            for (int t = 0; t < NCHUNK; ++t) {
                const float v = cs[lane + t * 32];
                if (v > lval) { lval = v; lidx = lane + t * 32; }
            }
        }
    }
}

extern "C" void kernel_launch(void* const* d_in, const int* in_sizes, int n_in,
                              void* d_out, int out_size, void* d_ws, size_t ws_size,
                              hipStream_t stream)
{
    const float*         q    = (const float*)d_in[0];
    const float*         g    = (const float*)d_in[1];
    const unsigned char* mask = (const unsigned char*)d_in[2];
    // d_in[3] is k (==32), fixed at compile time.

    float* ws_s = (float*)d_ws;
    int*   ws_i = (int*)((char*)d_ws + sizeof(float) * (size_t)BQ * NCHUNK * TOPK);

    int*   out_i = (int*)d_out;                       // best_i: B*K int32
    float* out_s = (float*)d_out + (size_t)BQ * TOPK; // best_s: B*K fp32

    const size_t smem = (size_t)16 * QS * sizeof(bf16) + (size_t)16 * SS * sizeof(float);
    hipFuncSetAttribute((const void*)sim_topk_phase1,
                        hipFuncAttributeMaxDynamicSharedMemorySize, (int)smem);

    // grid.x = m-tile so the 16 blocks sharing a 25MB gallery chunk are
    // dispatch-adjacent and reuse it through the 192MB L2.
    sim_topk_phase1<<<dim3(16, NCHUNK), 256, smem, stream>>>(q, g, mask, ws_s, ws_i);
    topk_merge<<<dim3(BQ), 32, 0, stream>>>(ws_s, ws_i, out_i, out_s);
}
